// _SpectralConv_31877247271510
// MI455X (gfx1250) — matrix operand order repaired and checked by reading, not compiled, not run on hardware
//
#include <hip/hip_runtime.h>
#include <math.h>
#include <stdint.h>

typedef __attribute__((ext_vector_type(16))) _Float16 v16h;
typedef __attribute__((ext_vector_type(8)))  float    v8f;

#define NB 4
#define NL 4096
#define NH 16
#define ND 64

#define NFFT  8192
#define NHALF 4096

// FIR taps: K = 145 (tile window then is a multiple of 32)
#define KTAPS 145
// Reversed extended filter table: hrev[i] = h_filt(191 - i); zero outside [0,KTAPS)
#define HEXT  224

#define TBLK  32            // output timesteps per block
#define WROWS 192           // x window rows staged per block (6 chunks of 32)
#define NCHUNK (WROWS / 32)
#define XW_STRIDE 196       // padded LDS row stride (f16 elems): 98 dwords -> 2-way banks max

// ---------------------------------------------------------------------------
// Kernel 1: per-head impulse response via direct inverse rDFT of the filter,
// stored REVERSED and in f16 so the FIR kernel's A-fragments are contiguous
// packed-f16 LDS loads (no cvt / half-swaps between WMMAs).
//   filt[f] = amp/(1+f^2)^(decay/2) * exp(-i*decay*atan(f))
//   h[m] = (1/N)[F0 + (-1)^m Re(F_{N/2}) + 2*sum_{f=1}^{N/2-1} mag*cos(phase + 2pi f m/N)]
// ---------------------------------------------------------------------------
__global__ void _spectral_build_filter(const float* __restrict__ log_amp,
                                       const float* __restrict__ log_decay,
                                       _Float16* __restrict__ hrev) {
    const int h = blockIdx.x;
    const int t = threadIdx.x;
    if (t >= HEXT) return;
    const int m = 191 - t;                                 // reversed index
    float val = 0.0f;
    if (m >= 0 && m < KTAPS) {
        const float amp   = log1pf(expf(log_amp[h]));
        const float decay = log1pf(expf(log_decay[h])) + 1e-4f;
        float acc = amp;                                   // f = 0 term
        {                                                  // f = N/2 term
            const float ff  = (float)NHALF;
            const float mag = amp * exp2f(-0.5f * decay * log2f(1.0f + ff * ff));
            const float ph  = -decay * atanf(ff);
            acc += ((m & 1) ? -1.0f : 1.0f) * mag * cosf(ph);
        }
        const float twopi = 6.28318530717958647692f;
        for (int f = 1; f < NHALF; ++f) {
            const float ff  = (float)f;
            const float mag = amp * exp2f(-0.5f * decay * log2f(1.0f + ff * ff));
            const float ph  = -decay * atanf(ff);
            const int   fr  = (f * m) & (NFFT - 1);        // exact mod-2pi reduction
            const float th  = twopi * ((float)fr * (1.0f / (float)NFFT));
            acc += 2.0f * mag * cosf(ph + th);
        }
        val = acc * (1.0f / (float)NFFT);
    }
    hrev[h * HEXT + t] = (_Float16)val;
}

// ---------------------------------------------------------------------------
// Kernel 2: causal FIR as WMMA f32_16x16x32_f16.
//   Block: (b, h, T0 = blockIdx.x*32). 8 waves, each one 16t x 16d tile.
//   All 12 fragments (6 A + 6 B) preloaded -> one dscnt wait -> 6 back-to-back
//   WMMAs on the accumulator chain.
// ---------------------------------------------------------------------------
__global__ __launch_bounds__(256, 2)
void _spectral_fir_wmma(const float* __restrict__ x,
                        const float* __restrict__ mix_logit,
                        const _Float16* __restrict__ hrev_g,
                        float* __restrict__ out) {
    __shared__ _Float16 xw[ND * XW_STRIDE];   // transposed window [d][j]
    __shared__ _Float16 hr[HEXT];             // reversed filter taps (f16)

    const int tid = threadIdx.x;
    const int b   = blockIdx.z;
    const int h   = blockIdx.y;
    const int T0  = blockIdx.x * TBLK;
    const int T0W = T0 - (WROWS - TBLK);      // window row j=0 -> time T0-160

    if (tid < HEXT) hr[tid] = hrev_g[h * HEXT + tid];

    // Stage x window into LDS as packed f16 pairs (one b32 store per thread/iter).
    // 256 threads cover 4 row-pairs x 64 d per iteration; coalesced 256B rows.
    for (int it = 0; it < WROWS / 8; ++it) {
        const int jp = it * 8 + (tid >> 6) * 2;           // even j
        const int d  = tid & 63;
        const int tg0 = T0W + jp;
        const int tg1 = tg0 + 1;
        float v0 = 0.0f, v1 = 0.0f;
        if (tg0 >= 0) v0 = x[(((size_t)b * NL + tg0) * NH + h) * ND + d];
        if (tg1 >= 0) v1 = x[(((size_t)b * NL + tg1) * NH + h) * ND + d];
        union { _Float16 hh[2]; uint32_t u; } pk;
        pk.hh[0] = (_Float16)v0;
        pk.hh[1] = (_Float16)v1;
        *(uint32_t*)&xw[d * XW_STRIDE + jp] = pk.u;       // 4B-aligned (stride even)
    }
    __syncthreads();

    const int wave  = tid >> 5;
    const int lane  = tid & 31;
    const int half  = lane >> 4;              // K-half selector (wave32 layouts)
    const int ln16  = lane & 15;              // M (A) / N (B,C,D)
    const int tt    = wave >> 2;              // t-tile 0..1
    const int dt    = wave & 3;               // d-tile 0..3
    const int obase = tt * 16;
    const int aBase = 31 - (obase + ln16);    // hrev base: A[e] = hr[aBase + 32c + kkA]
    const int bRow  = (dt * 16 + ln16) * XW_STRIDE;

    // Preload all fragments: DS loads batched, single wait, then pure WMMA chain.
    v16h afrag[NCHUNK], bfrag[NCHUNK];
    #pragma unroll
    for (int c = 0; c < NCHUNK; ++c) {
        #pragma unroll
        for (int e = 0; e < 16; ++e) {
            const int v = e >> 1, p = e & 1;
            // A 16x32 layout (matches B): K = e + 16*half
            const int kkA = 2 * v + p + 16 * half;
            afrag[c][e] = hr[aBase + 32 * c + kkA];        // contiguous pairs -> b32/b64 DS loads
            // B 32x16 layout (matches A): K = e + 16*half, N = lane%16
            bfrag[c][e] = xw[bRow + 32 * c + e + 16 * half];
        }
    }

    v8f acc = {0.f, 0.f, 0.f, 0.f, 0.f, 0.f, 0.f, 0.f};
    #pragma unroll
    for (int c = 0; c < NCHUNK; ++c) {
        acc = __builtin_amdgcn_wmma_f32_16x16x32_f16(
            /*neg_a=*/false, afrag[c], /*neg_b=*/false, bfrag[c],
            /*c_mod=*/(short)0, acc, /*reuse_a=*/false, /*reuse_b=*/false);
    }

    // Epilogue: out = mix*y + (1-mix)*x  (x residual reloaded in f32 for exactness).
    const float mix = 1.0f / (1.0f + expf(-mix_logit[h]));
    #pragma unroll
    for (int r = 0; r < 8; ++r) {
        const int t = T0 + obase + r + 8 * half;           // f32 C/D: M = r + 8*half
        const int d = dt * 16 + ln16;                      // N = lane%16
        const size_t idx = (((size_t)b * NL + t) * NH + h) * ND + d;
        const float xv = x[idx];
        out[idx] = mix * acc[r] + (1.0f - mix) * xv;
    }
}

extern "C" void kernel_launch(void* const* d_in, const int* in_sizes, int n_in,
                              void* d_out, int out_size, void* d_ws, size_t ws_size,
                              hipStream_t stream) {
    const float* x         = (const float*)d_in[0];
    const float* log_amp   = (const float*)d_in[1];
    const float* log_decay = (const float*)d_in[2];
    const float* mix_logit = (const float*)d_in[3];
    float* out  = (float*)d_out;
    _Float16* hrev = (_Float16*)d_ws;         // NH*HEXT f16 = 7 KB

    _spectral_build_filter<<<dim3(NH), dim3(256), 0, stream>>>(log_amp, log_decay, hrev);
    _spectral_fir_wmma<<<dim3(NL / TBLK, NH, NB), dim3(256), 0, stream>>>(x, mix_logit, hrev, out);
}
